// UnivNetLvcResidualBlock_64828236365885
// MI455X (gfx1250) — matrix operand order repaired and checked by reading, compiled, not run on hardware
//
#include <hip/hip_runtime.h>
#include <cstdint>

// ---------------------------------------------------------------------------
// UnivNet LVC residual block for MI455X (gfx1250, wave32, WMMA).
//
// Both the dilated conv (32ch -> 32ch, k=3, dil=3) and the location-variable
// conv (per-chunk 64x256 = [64 x 96] @ [96 x 256] GEMM) are executed on
// V_WMMA_F32_16X16X32_BF16: bf16 operands (full f32 exponent range),
// f32 accumulation. K = (tap, channel) = 3 K-steps of 32; each K-step's B
// operand is the activation tile shifted by the tap offset, so one LDS tile
// serves all taps.
//
// Workspace layout (requires ~24.6 MB):
//   kA    : bf16 [B][KL][t=3][mt=4][lane=32][slot=16]  (LVC A-fragments)
//   cwA   : bf16 [mt=2][t=3][lane=32][slot=16]         (conv A-fragments)
//   biasP : f32  [B][KL][64]                           (bias, transposed)
// ---------------------------------------------------------------------------

typedef __attribute__((ext_vector_type(16))) __bf16 v16bf;
typedef __attribute__((ext_vector_type(8)))  float  v8f;

#define DEVI __device__ __forceinline__

constexpr int Bn   = 16;
constexpr int Cn   = 32;       // HIDDEN
constexpr int Ln   = 32768;
constexpr int KLn  = 128;      // L / HOP
constexpr int OUTn = 64;

constexpr long long KA_ELEMS   = (long long)Bn * KLn * 3 * 4 * 512; // 12,582,912 bf16
constexpr long long CW_ELEMS   = 2 * 3 * 512;                       // 3072 bf16
constexpr long long BIAS_ELEMS = (long long)Bn * KLn * OUTn;        // 131072 f32

DEVI float leaky(float x) { return x >= 0.f ? x : 0.2f * x; }

// bf16 A-matrix (16x32) per-lane K index: K = 16*(v/4) + 8*(lane/16) + 2*(v%4) + j
DEVI int a_frag_k(int lane, int slot) {
    int v = slot >> 1, j = slot & 1;
    return 16 * (v >> 2) + 8 * (lane >> 4) + 2 * (v & 3) + j;
}

// ---------------------------------------------------------------------------
// Repack: gather LVC kernels / conv_w into WMMA A-fragment order (bf16),
// transpose bias. Writes are fully coalesced; scattered reads hit L2
// (whole kernel tensor = 50 MB << 192 MB L2).
// ---------------------------------------------------------------------------
__global__ __launch_bounds__(256) void repack_kernel(
    const float* __restrict__ kern,   // (B,32,64,3,KL)
    const float* __restrict__ convw,  // (32,32,3)
    const float* __restrict__ bias,   // (B,64,KL)
    __bf16* __restrict__ kA,
    __bf16* __restrict__ cwA,
    float*  __restrict__ biasP)
{
    const long long N0 = KA_ELEMS;
    const long long N1 = N0 + CW_ELEMS;
    const long long N2 = N1 + BIAS_ELEMS;
    const long long stride = (long long)gridDim.x * blockDim.x;
    for (long long e = (long long)blockIdx.x * blockDim.x + threadIdx.x;
         e < N2; e += stride) {
        if (e < N0) {
            int slot = (int)(e & 15);
            int lane = (int)((e >> 4) & 31);
            int mt   = (int)((e >> 9) & 3);
            long long r = e >> 11;
            int t  = (int)(r % 3);
            long long q = r / 3;
            int kl = (int)(q & 127);
            int b  = (int)(q >> 7);
            int i = a_frag_k(lane, slot);            // channel (K within step)
            int o = mt * 16 + (lane & 15);           // output channel (M)
            float val = kern[(((long long)(b * Cn + i) * OUTn + o) * 3 + t) * KLn + kl];
            kA[e] = (__bf16)val;
        } else if (e < N1) {
            long long e2 = e - N0;
            int mt   = (int)(e2 / 1536);
            int rem  = (int)(e2 % 1536);
            int t    = rem / 512;
            int lane = (rem >> 4) & 31;
            int slot = rem & 15;
            int i = a_frag_k(lane, slot);
            int o = mt * 16 + (lane & 15);
            cwA[e2] = (__bf16)convw[(o * Cn + i) * 3 + t];
        } else {
            long long e3 = e - N1;
            int o  = (int)(e3 & 63);
            int kl = (int)((e3 >> 6) & 127);
            int b  = (int)(e3 >> 13);
            biasP[e3] = bias[((long long)(b * OUTn + o)) * KLn + kl];
        }
    }
}

// ---------------------------------------------------------------------------
// Fused main kernel: one block per (batch, chunk).
//   sXin : leaky(h) halo tile, [s=280][i=32] bf16, s=0 <-> l = kl*256 - 11
//   sX2  : conv output tile,   [s=272][i=32] bf16, s=0 <-> l = kl*256 - 8
// ---------------------------------------------------------------------------
#define XPITCH 34   // row pitch (elements); even -> dword-aligned pair loads

__global__ __launch_bounds__(256) void lvc_block_kernel(
    const float*  __restrict__ hid,    // (B,32,L)
    const float*  __restrict__ convb,  // (32,)
    const __bf16* __restrict__ kA,
    const __bf16* __restrict__ cwA,
    const float*  __restrict__ biasP,
    float* __restrict__ outp)          // (B,32,L)
{
    __shared__ __bf16 sXin[280 * XPITCH];
    __shared__ __bf16 sX2 [272 * XPITCH];
    __shared__ float  sBias[64];
    __shared__ float  sCB[32];

    const int blk  = blockIdx.x;          // b*128 + kl
    const int b    = blk >> 7;
    const int kl   = blk & 127;
    const int tid  = threadIdx.x;
    const int wave = tid >> 5;
    const int lane = tid & 31;

    const __bf16* kAb = kA + (long long)blk * 6144;
    // warm L2/WGP$ for this block's A panel (global_prefetch_b8)
    __builtin_prefetch(kAb + tid * 24, 0, 1);

    if (tid < 64) sBias[tid] = biasP[((long long)blk << 6) + tid];
    if (tid < 32) sCB[tid]   = convb[tid];

    // ---- Phase 1: load h halo tile, apply leaky, store bf16 [s][i] --------
    for (int idx = tid; idx < 280 * Cn; idx += 256) {
        int i = idx / 280, s = idx % 280;
        int l = (kl << 8) + s - 11;
        float v = 0.f;
        if (l >= 0 && l < Ln) v = hid[((long long)(b * Cn + i)) * Ln + l];
        sXin[s * XPITCH + i] = (__bf16)leaky(v);
    }
    __syncthreads();

    // conv A fragments (shared across blocks; served by cache)
    v16bf cwF[2][3];
#pragma unroll
    for (int mt = 0; mt < 2; ++mt)
#pragma unroll
        for (int t = 0; t < 3; ++t)
            cwF[mt][t] = *(const v16bf*)(cwA + ((mt * 3 + t) * 32 + lane) * 16);

    const int nlo  = lane & 15;        // N index within tile
    const int half = lane >> 4;
    const int mh   = half * 8;         // D-matrix M base for this lane half
    const int ih   = half * 8;         // B-matrix K base for this lane half

    // ---- Phase 2: dilated conv via WMMA; 17 N-tiles cover s2 in [0,272) ---
    for (int nt = wave; nt < 17; nt += 8) {
        int n0 = nt << 4;
        v8f acc0 = {0,0,0,0,0,0,0,0};
        v8f acc1 = {0,0,0,0,0,0,0,0};
#pragma unroll
        for (int t = 0; t < 3; ++t) {
            // B[(i),n] = xin[i, s2 + 3*(t-1)] ; xin col = s2 + 3*t (coord shift +3)
            union { v16bf v; unsigned u[8]; } bf;
            int col = n0 + nlo + 3 * t;
#pragma unroll
            for (int vv = 0; vv < 8; ++vv)
                bf.u[vv] = *(const unsigned*)&sXin[col * XPITCH + ih + 2 * vv + 8 * (vv >> 2)];
            acc0 = __builtin_amdgcn_wmma_f32_16x16x32_bf16(
                false, cwF[0][t], false, bf.v, (short)0, acc0, false, false);
            acc1 = __builtin_amdgcn_wmma_f32_16x16x32_bf16(
                false, cwF[1][t], false, bf.v, (short)0, acc1, false, false);
        }
        int s2 = n0 + nlo;
#pragma unroll
        for (int r = 0; r < 8; ++r) {
            int o0 = mh + r;
            int o1 = 16 + mh + r;
            sX2[s2 * XPITCH + o0] = (__bf16)leaky(acc0[r] + sCB[o0]);
            sX2[s2 * XPITCH + o1] = (__bf16)leaky(acc1[r] + sCB[o1]);
        }
    }
    __syncthreads();
    // LVC zero-padding at sequence edges (reference pads x2, not h)
    if (kl == 0       && tid < Cn) sX2[7   * XPITCH + tid] = (__bf16)0.f;  // l = -1
    if (kl == KLn - 1 && tid < Cn) sX2[264 * XPITCH + tid] = (__bf16)0.f;  // l = L
    __syncthreads();

    // ---- Phase 3: LVC GEMM (64x256 = [64x96]@[96x256]) + gate + residual --
    for (int nt = wave; nt < 16; nt += 8) {
        int n0 = nt << 4;
        v8f z = {0,0,0,0,0,0,0,0};
        v8f acc[4] = {z, z, z, z};
#pragma unroll
        for (int t = 0; t < 3; ++t) {
            // B[(i),n] = x2[i, s + t - 1] ; x2 col = s + t + 7 (coord shift +8)
            union { v16bf v; unsigned u[8]; } bf;
            int col = n0 + nlo + t + 7;
#pragma unroll
            for (int vv = 0; vv < 8; ++vv)
                bf.u[vv] = *(const unsigned*)&sX2[col * XPITCH + ih + 2 * vv + 8 * (vv >> 2)];
#pragma unroll
            for (int mt = 0; mt < 4; ++mt) {
                v16bf a = *(const v16bf*)(kAb + ((t * 4 + mt) * 32 + lane) * 16);
                acc[mt] = __builtin_amdgcn_wmma_f32_16x16x32_bf16(
                    false, a, false, bf.v, (short)0, acc[mt], false, false);
            }
        }
        int l = (kl << 8) + n0 + nlo;
#pragma unroll
        for (int mt = 0; mt < 2; ++mt) {
#pragma unroll
            for (int r = 0; r < 8; ++r) {
                int c = mt * 16 + mh + r;                 // channel 0..31
                float av = acc[mt][r]     + sBias[c];      // sigmoid half
                float gv = acc[mt + 2][r] + sBias[32 + c]; // tanh half
                float sig = 1.f / (1.f + __expf(-av));
                float y   = sig * tanhf(gv);
                long long off = ((long long)(b * Cn + c)) * Ln + l;
                outp[off] = hid[off] + y;
            }
        }
    }
}

// ---------------------------------------------------------------------------
extern "C" void kernel_launch(void* const* d_in, const int* in_sizes, int n_in,
                              void* d_out, int out_size, void* d_ws, size_t ws_size,
                              hipStream_t stream) {
    (void)in_sizes; (void)n_in; (void)out_size;
    const float* hid   = (const float*)d_in[0];  // (16,32,32768)
    const float* kern  = (const float*)d_in[1];  // (16,32,64,3,128)
    const float* bias  = (const float*)d_in[2];  // (16,64,128)
    const float* convw = (const float*)d_in[3];  // (32,32,3)
    const float* convb = (const float*)d_in[4];  // (32,)
    // d_in[5] = hop_size (compile-time constant 256)

    __bf16* kA    = (__bf16*)d_ws;
    __bf16* cwA   = kA + KA_ELEMS;
    float*  biasP = (float*)(cwA + CW_ELEMS);   // byte offset multiple of 4
    (void)ws_size;  // requires ~24.6 MB workspace

    repack_kernel<<<4096, 256, 0, stream>>>(kern, convw, bias, kA, cwA, biasP);
    lvc_block_kernel<<<Bn * KLn, 256, 0, stream>>>(hid, convb, kA, cwA, biasP,
                                                   (float*)d_out);
}
